// AdjacentAttention_11407433138537
// MI455X (gfx1250) — matrix operand order repaired and checked by reading, not compiled, not run on hardware
//
#include <hip/hip_runtime.h>
#include <hip/hip_bf16.h>
#include <math.h>

#define HEADS      4
#define DIM_HEAD   64
#define DIM        256
#define INNER      256
#define QKV_COLS   768
#define ATT_SCALE  0.125f   // 64^-0.5

typedef __bf16 bf16_t;
typedef __attribute__((ext_vector_type(16))) __bf16 v16bf;
typedef __attribute__((ext_vector_type(8)))  __bf16 v8bf;
typedef __attribute__((ext_vector_type(8)))  float  v8f;

// ---------------------------------------------------------------------------
// fp32 -> bf16 conversion (RNE via clang cast)
// ---------------------------------------------------------------------------
__global__ void f32_to_bf16_kernel(const float* __restrict__ in,
                                   bf16_t* __restrict__ out,
                                   long long cnt) {
  long long t = (long long)blockIdx.x * blockDim.x + threadIdx.x;
  if (t < cnt) out[t] = (bf16_t)in[t];
}

// ---------------------------------------------------------------------------
// Fragment loaders per CDNA5 ISA 7.12.2 (wave32):
//   A 16x32 : lane&15 = M row; half=lane>>4; elems 0..7 -> K = k0+8*half..,
//             elems 8..15 -> K = k0+16+8*half  (two contiguous 16B loads)
//   B 32x16 : lane&15 = N col; elems use same K order as A (strided gathers)
//   C 16x16 : vgpr r -> M = r + 8*half, N = lane&15
// ---------------------------------------------------------------------------
__device__ __forceinline__ v16bf load_a_frag(const bf16_t* arow_ptr, int k0, int half) {
  v8bf lo = *(const v8bf*)(arow_ptr + k0 + 8 * half);
  v8bf hi = *(const v8bf*)(arow_ptr + k0 + 16 + 8 * half);
  return __builtin_shufflevector(lo, hi, 0,1,2,3,4,5,6,7,8,9,10,11,12,13,14,15);
}

__device__ __forceinline__ v16bf load_b_frag(const bf16_t* b_krow_n, int N, int half) {
  v16bf r;
#pragma unroll
  for (int s = 0; s < 8; ++s) {
    r[s] = b_krow_n[(size_t)(8 * half + s) * N];
    r[s + 8] = b_krow_n[(size_t)(16 + 8 * half + s) * N];
  }
  return r;
}

#define WMMA_BF16(A_, B_, C_) \
  __builtin_amdgcn_wmma_f32_16x16x32_bf16(false, (A_), false, (B_), (short)0, (C_), false, false)

// ---------------------------------------------------------------------------
// Dense GEMM: C[M,N] = A[M,K](bf16,rm) * B[K,N](bf16,rm) (+ bias)
// One wave32 computes a 32x32 C macro-tile: 2 A-frags x 2 B-frags ->
// 4 accumulators, 4 v_wmma per 32-wide K step (32B fragment loads per WMMA).
// ---------------------------------------------------------------------------
__global__ void wmma_gemm_bf16_kernel(const bf16_t* __restrict__ A,
                                      const bf16_t* __restrict__ B,
                                      float* __restrict__ C,
                                      const float* __restrict__ bias,
                                      int M, int N, int K) {
  const int wave = threadIdx.x >> 5;
  const int lane = threadIdx.x & 31;
  const int half = lane >> 4;
  const int m0 = blockIdx.x << 5;                                   // 32 rows
  const int n0 = (blockIdx.y * (blockDim.x >> 5) + wave) << 5;      // 32 cols
  if (m0 >= M || n0 >= N) return;

  int ar0 = m0 + (lane & 15);
  int ar1 = m0 + 16 + (lane & 15);
  if (ar0 >= M) ar0 = M - 1;                   // clamp (M is 32-aligned here)
  if (ar1 >= M) ar1 = M - 1;
  const bf16_t* aptr0 = A + (size_t)ar0 * K;
  const bf16_t* aptr1 = A + (size_t)ar1 * K;

  v8f c00 = {}, c01 = {}, c10 = {}, c11 = {};
  for (int k0 = 0; k0 < K; k0 += 32) {
    v16bf a0 = load_a_frag(aptr0, k0, half);
    v16bf a1 = load_a_frag(aptr1, k0, half);
    const bf16_t* bp = B + (size_t)k0 * N + n0 + (lane & 15);
    v16bf b0 = load_b_frag(bp, N, half);
    v16bf b1 = load_b_frag(bp + 16, N, half);
    c00 = WMMA_BF16(a0, b0, c00);
    c01 = WMMA_BF16(a0, b1, c01);
    c10 = WMMA_BF16(a1, b0, c10);
    c11 = WMMA_BF16(a1, b1, c11);
  }

  const int colL = n0 + (lane & 15);
  const int colR = colL + 16;
  const float bL = bias ? bias[colL] : 0.0f;
  const float bR = bias ? bias[colR] : 0.0f;
#pragma unroll
  for (int r = 0; r < 8; ++r) {
    int row0 = m0 + r + 8 * half;
    int row1 = row0 + 16;
    if (row0 < M) {
      C[(size_t)row0 * N + colL] = c00[r] + bL;
      C[(size_t)row0 * N + colR] = c01[r] + bR;
    }
    if (row1 < M) {
      C[(size_t)row1 * N + colL] = c10[r] + bL;
      C[(size_t)row1 * N + colR] = c11[r] + bR;
    }
  }
}

// ---------------------------------------------------------------------------
// Gather attention with online softmax.
// One wave32 per (token, head); lane owns dims {2*lane, 2*lane+1} of 64.
// qkv layout per row (768 f32): [ q(4x64) | k(4x64) | v(4x64) ]
// Null token (always unmasked) first, then the 32 gathered neighbors
// (k/v reads are L2-resident: whole k/v block is ~41MB < 192MB L2).
// ---------------------------------------------------------------------------
__device__ __forceinline__ float wave_sum32(float v) {
#pragma unroll
  for (int off = 16; off > 0; off >>= 1) v += __shfl_xor(v, off, 32);
  return v;
}

__global__ void attn_gather_kernel(const float* __restrict__ qkv,
                                   const long long* __restrict__ adj,
                                   const unsigned char* __restrict__ mask,
                                   const float* __restrict__ null_k,
                                   const float* __restrict__ null_v,
                                   bf16_t* __restrict__ out_bf,
                                   int n, int a) {
  const int wave = threadIdx.x >> 5;
  const int lane = threadIdx.x & 31;
  const long long p = (long long)blockIdx.x * (blockDim.x >> 5) + wave;
  if (p >= (long long)n * HEADS) return;
  const int i = (int)(p >> 2);   // token
  const int h = (int)(p & 3);    // head
  const int d0 = lane * 2;

  const float* qrow = qkv + (size_t)i * QKV_COLS + h * DIM_HEAD;
  const float q0 = qrow[d0] * ATT_SCALE;
  const float q1 = qrow[d0 + 1] * ATT_SCALE;

  float m = -3.0e38f, l = 0.0f, acc0 = 0.0f, acc1 = 0.0f;

  // j = 0 : learned null token
  {
    float2 kk = *(const float2*)(null_k + h * DIM_HEAD + d0);
    float  s  = wave_sum32(q0 * kk.x + q1 * kk.y);
    float  mn = fmaxf(m, s);
    float  co = __expf(m - mn);
    float  pw = __expf(s - mn);
    float2 vv = *(const float2*)(null_v + h * DIM_HEAD + d0);
    l    = l * co + pw;
    acc0 = acc0 * co + pw * vv.x;
    acc1 = acc1 * co + pw * vv.y;
    m    = mn;
  }

  const long long*     arow = adj  + (size_t)i * a;
  const unsigned char* mrow = mask + (size_t)i * a;
  for (int j = 0; j < a; ++j) {
    if (!mrow[j]) continue;                        // wave-uniform branch
    const int idx = (int)arow[j];
    const float* base = qkv + (size_t)idx * QKV_COLS + h * DIM_HEAD;
    float2 kk = *(const float2*)(base + INNER + d0);        // k
    float  s  = wave_sum32(q0 * kk.x + q1 * kk.y);
    float  mn = fmaxf(m, s);
    float  co = __expf(m - mn);
    float  pw = __expf(s - mn);
    float2 vv = *(const float2*)(base + 2 * INNER + d0);    // v
    l    = l * co + pw;
    acc0 = acc0 * co + pw * vv.x;
    acc1 = acc1 * co + pw * vv.y;
    m    = mn;
  }

  const float inv = 1.0f / l;
  bf16_t* o = out_bf + (size_t)i * INNER + h * DIM_HEAD + d0;
  o[0] = (bf16_t)(acc0 * inv);
  o[1] = (bf16_t)(acc1 * inv);
}

// ---------------------------------------------------------------------------
// Launch
// ---------------------------------------------------------------------------
static inline size_t align_up(size_t x, size_t a) { return (x + a - 1) & ~(a - 1); }

extern "C" void kernel_launch(void* const* d_in, const int* in_sizes, int n_in,
                              void* d_out, int out_size, void* d_ws, size_t ws_size,
                              hipStream_t stream) {
  const float*         x      = (const float*)d_in[0];
  const long long*     adj    = (const long long*)d_in[1];
  const unsigned char* mask   = (const unsigned char*)d_in[2];
  const float*         w_qkv  = (const float*)d_in[3];
  const float*         w_out  = (const float*)d_in[4];
  const float*         b_out  = (const float*)d_in[5];
  const float*         null_k = (const float*)d_in[6];
  const float*         null_v = (const float*)d_in[7];
  float*               out    = (float*)d_out;

  const int n = in_sizes[0] / DIM;      // 20000
  const int a = in_sizes[1] / n;        // 32

  // workspace layout
  char* ws = (char*)d_ws;
  bf16_t* x_bf = (bf16_t*)ws;    ws += align_up((size_t)n * DIM * sizeof(bf16_t), 256);
  bf16_t* wqkv_bf = (bf16_t*)ws; ws += align_up((size_t)DIM * QKV_COLS * sizeof(bf16_t), 256);
  bf16_t* wout_bf = (bf16_t*)ws; ws += align_up((size_t)INNER * DIM * sizeof(bf16_t), 256);
  float*  qkv_f32 = (float*)ws;  ws += align_up((size_t)n * QKV_COLS * sizeof(float), 256);
  bf16_t* att_bf  = (bf16_t*)ws; ws += align_up((size_t)n * INNER * sizeof(bf16_t), 256);

  // 1) fp32 -> bf16 conversions
  {
    long long c1 = (long long)n * DIM;
    f32_to_bf16_kernel<<<(unsigned)((c1 + 255) / 256), 256, 0, stream>>>(x, x_bf, c1);
    long long c2 = (long long)DIM * QKV_COLS;
    f32_to_bf16_kernel<<<(unsigned)((c2 + 255) / 256), 256, 0, stream>>>(w_qkv, wqkv_bf, c2);
    long long c3 = (long long)INNER * DIM;
    f32_to_bf16_kernel<<<(unsigned)((c3 + 255) / 256), 256, 0, stream>>>(w_out, wout_bf, c3);
  }

  const int mtiles32 = (n + 31) / 32;   // 625
  // 2) QKV projection: [n,256] x [256,768] -> qkv_f32 (WMMA bf16, 32x32/wave)
  {
    dim3 grid(mtiles32, (QKV_COLS + 32 * 8 - 1) / (32 * 8));   // 625 x 3
    wmma_gemm_bf16_kernel<<<grid, 256, 0, stream>>>(
        x_bf, wqkv_bf, qkv_f32, nullptr, n, QKV_COLS, DIM);
  }

  // 3) gather attention + online softmax -> att_bf [n, 256]
  {
    long long pairs = (long long)n * HEADS;
    unsigned blocks = (unsigned)((pairs + 7) / 8);              // 8 waves/block
    attn_gather_kernel<<<blocks, 256, 0, stream>>>(
        qkv_f32, adj, mask, null_k, null_v, att_bf, n, a);
  }

  // 4) output projection + bias: [n,256] x [256,256] + b_out -> d_out (f32)
  {
    dim3 grid(mtiles32, (DIM + 32 * 8 - 1) / (32 * 8));         // 625 x 1
    wmma_gemm_bf16_kernel<<<grid, 256, 0, stream>>>(
        att_bf, wout_bf, out, b_out, n, DIM, INNER);
  }
}
